// DMamba_16801912062578
// MI455X (gfx1250) — matrix operand order repaired and checked by reading, not compiled, not run on hardware
//
#include <hip/hip_runtime.h>
#include <hip/hip_bf16.h>
#include <math.h>

// ---------------------------------------------------------------------------
// DMamba for MI455X (gfx1250):
//  - GEMM/conv on v_wmma_f32_16x16x32_f16, 16x64 per-wave macro-tiles.
//  - B tiles staged through LDS once per block (4-wave broadcast, fragment-
//    ready f16 layout -> ds_load_b128 reads), A streamed + prefetched.
//  - Selective scan double-buffers B_t/C_t via GLOBAL_LOAD_ASYNC_TO_LDS_B128
//    + s_wait_asynccnt (ASYNCcnt), state register-resident.
// ---------------------------------------------------------------------------

typedef __attribute__((ext_vector_type(16))) _Float16 v16h;
typedef __attribute__((ext_vector_type(8)))  float    v8f;

constexpr int B_   = 16;
constexpr int T_   = 30;
constexpr int NE_  = 512;
constexpr int DM_  = 512;
constexpr int DIN_ = 1024;
constexpr int NS_  = 64;     // state dim N
constexpr int DTR_ = 32;
constexpr int L_   = 3;
constexpr int V_   = 18;
constexpr int SEQ_ = 3 * T_;        // 90
constexpr int ROWS_ = B_ * SEQ_;    // 1440
constexpr int BT_  = B_ * T_;       // 480

#define ACT_NONE     0
#define ACT_RELU     1
#define ACT_TANH     2
#define ACT_SOFTPLUS 3

__device__ __forceinline__ float softplusf(float v) {
  return (v > 20.f) ? v : log1pf(expf(v));
}

// ---------------------------------------------------------------------------
// Generic GEMM: C[M,N] = act(A[M,K] @ B[K,N] + bias) (+ C residual)
// Block = 4 waves stacked along M (64x64 block tile). Per K-step the 32x64
// B tile is cooperatively staged into LDS in fragment layout (f16), then all
// 4 waves issue 4 WMMAs each against one reused A fragment.
// Fragment striping used here (contiguous K, same for A and B):
//   A lane l, elem j -> K = ((l>>4)<<4) + j,             M row = l&15
//   B lane l, elem j -> K = ((l>>4)<<4) + j,             N col = l&15
//   C lane l, vgpr r -> M row = r + ((l>>4)<<3),         N col = l&15
// NOTE: no wave-level early exit -- barriers inside the K loop require all
// waves to participate; tail waves run with mv=false and guarded stores.
// ---------------------------------------------------------------------------
template <int ACT, bool BIAS, bool RESID>
__global__ void gemm_wmma(const float* __restrict__ A, int lda,
                          const float* __restrict__ Bm, int ldb,
                          float* __restrict__ C, int ldc,
                          const float* __restrict__ bias,
                          int M, int N, int K) {
  const int lane = threadIdx.x;             // 0..31
  const int wave = threadIdx.y;             // 0..3
  const int tid  = wave * 32 + lane;        // 0..127
  const int m0   = (blockIdx.y * 4 + wave) * 16;
  const int n0   = blockIdx.x * 64;
  const int mr   = m0 + (lane & 15);
  const int ncL  = lane & 15;
  const int kA   = (lane >> 4) << 4;
  const bool mv  = (mr < M);
  const float* __restrict__ Arow = A + (size_t)mr * lda;

  __shared__ __align__(32) _Float16 bsh[4][32][16];   // [nt][lane][j]

  v8f acc[4] = {};
#pragma unroll 2
  for (int k0 = 0; k0 < K; k0 += 32) {
    // ---- cooperative B-tile stage: 32(K) x 64(N) -> fragment layout ----
    __syncthreads();                        // protect previous tile
#pragma unroll
    for (int i = 0; i < 16; ++i) {
      const int flat = i * 128 + tid;       // 0..2047
      const int kk = flat >> 6;             // 0..31
      const int nn = flat & 63;             // 0..63
      const int n = n0 + nn;
      const float v = (n < N) ? Bm[(size_t)(k0 + kk) * ldb + n] : 0.f;
      bsh[nn >> 4][((kk >> 4) << 4) | (nn & 15)][kk & 15] = (_Float16)v;
    }
    __syncthreads();

    // ---- A fragment (two contiguous 32B chunks per lane) ----
    v16h a;
#pragma unroll
    for (int j = 0; j < 16; ++j) {
      const int ka = k0 + kA + j;
      a[j] = (_Float16)(mv ? Arow[ka] : 0.f);
    }
    if (mv) __builtin_prefetch(Arow + k0 + 96, 0, 1);  // global_prefetch_b8

#pragma unroll
    for (int nt = 0; nt < 4; ++nt) {
      const v16h b = *(const v16h*)(&bsh[nt][lane][0]);  // ds_load_b128 x2
      acc[nt] = __builtin_amdgcn_wmma_f32_16x16x32_f16(false, a, false, b,
                                                       (short)0, acc[nt],
                                                       false, false);
    }
  }

  const int mbase = m0 + ((lane >> 4) << 3);
#pragma unroll
  for (int nt = 0; nt < 4; ++nt) {
    const int nc = n0 + nt * 16 + ncL;
    if (nc >= N) continue;
    const float bv = BIAS ? bias[nc] : 0.f;
#pragma unroll
    for (int r = 0; r < 8; ++r) {
      const int m = mbase + r;
      if (m < M) {
        float v = acc[nt][r] + bv;
        if (ACT == ACT_RELU)     v = v > 0.f ? v : 0.f;
        if (ACT == ACT_TANH)     v = tanhf(v);
        if (ACT == ACT_SOFTPLUS) v = softplusf(v);
        if (RESID) v += C[(size_t)m * ldc + nc];
        C[(size_t)m * ldc + nc] = v;
      }
    }
  }
}

// ---------------------------------------------------------------------------
// Implicit-GEMM conv (NCHW, VALID) + bias + ReLU on WMMA.
// M = NIMG*OH*OW rows, N = OC = 16*NT, K = IC*KW*KW. Weights OIHW (N x K).
// Same LDS B-staging as gemm_wmma (weights broadcast to all 4 waves).
// ---------------------------------------------------------------------------
template <int NT>
__global__ void conv_wmma(const float* __restrict__ X, const float* __restrict__ W,
                          const float* __restrict__ bias, float* __restrict__ Y,
                          int IC, int IH, int IW, int OC, int OH, int OW,
                          int KW, int stride, int K, int M) {
  const int lane = threadIdx.x;
  const int wave = threadIdx.y;
  const int tid  = wave * 32 + lane;
  const int m0   = (blockIdx.y * 4 + wave) * 16;
  const int mr   = m0 + (lane & 15);
  const bool mv  = (mr < M);
  int img = 0, oy = 0, ox = 0;
  if (mv) {
    img = mr / (OH * OW);
    const int rr = mr % (OH * OW);
    oy = rr / OW; ox = rr % OW;
  }
  const int ncL = lane & 15;
  const int kA  = (lane >> 4) << 4;
  const int KK  = KW * KW;
  const int NE16 = NT * 16;                 // elements per K row staged

  __shared__ __align__(32) _Float16 bsh[NT][32][16];

  v8f acc[NT] = {};
#pragma unroll 2
  for (int k0 = 0; k0 < K; k0 += 32) {
    __syncthreads();
    // stage 32(K) x OC(N) weight tile; W layout is [n*K + k]
    for (int flat = tid; flat < 32 * NE16; flat += 128) {
      const int kk = flat >> (NT == 4 ? 6 : 5);
      const int nn = flat & (NE16 - 1);
      const float v = W[(size_t)nn * K + (k0 + kk)];
      bsh[nn >> 4][((kk >> 4) << 4) | (nn & 15)][kk & 15] = (_Float16)v;
    }
    __syncthreads();

    v16h a;
#pragma unroll
    for (int j = 0; j < 16; ++j) {
      const int ka = k0 + kA + j;
      float av = 0.f;
      if (mv) {
        const int c = ka / KK, rr = ka % KK;
        const int ky = rr / KW, kx = rr % KW;
        av = X[(size_t)((img * IC + c) * IH + oy * stride + ky) * IW + ox * stride + kx];
      }
      a[j] = (_Float16)av;
    }
#pragma unroll
    for (int nt = 0; nt < NT; ++nt) {
      const v16h b = *(const v16h*)(&bsh[nt][lane][0]);
      acc[nt] = __builtin_amdgcn_wmma_f32_16x16x32_f16(false, a, false, b,
                                                       (short)0, acc[nt],
                                                       false, false);
    }
  }

  const int mbase = m0 + ((lane >> 4) << 3);
#pragma unroll
  for (int nt = 0; nt < NT; ++nt) {
    const int nc = nt * 16 + ncL;
    const float bv = bias[nc];
#pragma unroll
    for (int r = 0; r < 8; ++r) {
      const int m = mbase + r;
      if (m < M) {
        float v = acc[nt][r] + bv;
        v = v > 0.f ? v : 0.f;
        const int i2 = m / (OH * OW), rr = m % (OH * OW);
        const int y2 = rr / OW, x2 = rr % OW;
        Y[(size_t)((i2 * OC + nc) * OH + y2) * OW + x2] = v;
      }
    }
  }
}

// ---------------------------------------------------------------------------
// Token assembly: [rtg_e, state_e, act_e] interleaved per timestep.
// ---------------------------------------------------------------------------
__global__ void build_tokens(const float* __restrict__ rtgs, const int* __restrict__ actions,
                             const float* __restrict__ state_e,
                             const float* __restrict__ ret_w, const float* __restrict__ ret_b,
                             const float* __restrict__ act_emb, float* __restrict__ tok) {
  const int idx = blockIdx.x * blockDim.x + threadIdx.x;
  if (idx >= ROWS_ * NE_) return;
  const int c = idx & (NE_ - 1);
  const int row = idx >> 9;                 // NE_=512
  const int b = row / SEQ_, s = row % SEQ_;
  const int t = s / 3, kind = s % 3;
  float v;
  if (kind == 0)      v = tanhf(rtgs[b * T_ + t] * ret_w[c] + ret_b[c]);
  else if (kind == 1) v = state_e[(size_t)(b * T_ + t) * NE_ + c];
  else {
    const int a = actions[b * T_ + t];
    v = tanhf(act_emb[(size_t)a * NE_ + c]);
  }
  tok[idx] = v;
}

// ---------------------------------------------------------------------------
// RMSNorm over rows of 512.
// ---------------------------------------------------------------------------
__global__ void rmsnorm_k(const float* __restrict__ x, const float* __restrict__ w,
                          float* __restrict__ y) {
  const int row = blockIdx.x;
  __shared__ float red[128];
  float s = 0.f;
  for (int c = threadIdx.x; c < DM_; c += 128) {
    const float v = x[(size_t)row * DM_ + c];
    s += v * v;
  }
  red[threadIdx.x] = s;
  __syncthreads();
  for (int off = 64; off > 0; off >>= 1) {
    if (threadIdx.x < off) red[threadIdx.x] += red[threadIdx.x + off];
    __syncthreads();
  }
  const float inv = rsqrtf(red[0] / (float)DM_ + 1e-5f);
  for (int c = threadIdx.x; c < DM_; c += 128)
    y[(size_t)row * DM_ + c] = x[(size_t)row * DM_ + c] * inv * w[c];
}

// ---------------------------------------------------------------------------
// Causal depthwise conv1d (k=4, left pad 3) + SiLU. Input = xs half of xz.
// ---------------------------------------------------------------------------
__global__ void dwconv_silu(const float* __restrict__ xz, const float* __restrict__ cw,
                            const float* __restrict__ cb, float* __restrict__ xs) {
  const int idx = blockIdx.x * blockDim.x + threadIdx.x;
  if (idx >= ROWS_ * DIN_) return;
  const int d = idx & (DIN_ - 1);
  const int row = idx >> 10;                // DIN_=1024
  const int b = row / SEQ_, t = row % SEQ_;
  float acc = cb[d];
#pragma unroll
  for (int k = 0; k < 4; ++k) {
    const int tt = t - 3 + k;
    if (tt >= 0)
      acc += cw[d * 4 + k] * xz[(size_t)(b * SEQ_ + tt) * (2 * DIN_) + d];
  }
  xs[idx] = acc / (1.f + expf(-acc));       // silu
}

// ---------------------------------------------------------------------------
// Selective scan: one lane per (b,d); 64 states + 64 A values in registers.
// B_t/C_t are 128 contiguous floats of x_dbl -> double-buffered into LDS by
// wave 0 via GLOBAL_LOAD_ASYNC_TO_LDS_B128 (ASYNCcnt), overlapping the next
// timestep's fetch with the current 64-wide state update.
// ---------------------------------------------------------------------------
__global__ void sscan(const float* __restrict__ xs, const float* __restrict__ delta,
                      const float* __restrict__ xdbl, const float* __restrict__ alog,
                      const float* __restrict__ Dp, float* __restrict__ y) {
  const int b   = blockIdx.y;
  const int tid = threadIdx.x;              // 0..127
  const int d   = blockIdx.x * 128 + tid;
  __shared__ float BC[2][128];              // [0..63]=B_t, [64..127]=C_t
  float A_[NS_], st[NS_];
#pragma unroll
  for (int n = 0; n < NS_; ++n) {
    A_[n] = -expf(alog[(size_t)d * NS_ + n]);
    st[n] = 0.f;
  }
  const float Dd = Dp[d];
  const int lane = tid & 31;

  if (tid < 32) {                           // prologue: fetch t=0 into BC[0]
    const unsigned ldsOff = (unsigned)(size_t)&BC[0][lane * 4];
    const float* g = xdbl + (size_t)(b * SEQ_) * 160 + DTR_ + lane * 4;
    asm volatile("global_load_async_to_lds_b128 %0, %1, off"
                 :: "v"(ldsOff), "v"(g) : "memory");
  }

  for (int t = 0; t < SEQ_; ++t) {
    if (tid < 32) {
      if (t + 1 < SEQ_) {                   // prefetch t+1 into other buffer
        const unsigned ldsOff = (unsigned)(size_t)&BC[(t + 1) & 1][lane * 4];
        const float* g = xdbl + (size_t)(b * SEQ_ + t + 1) * 160 + DTR_ + lane * 4;
        asm volatile("global_load_async_to_lds_b128 %0, %1, off"
                     :: "v"(ldsOff), "v"(g) : "memory");
        asm volatile("s_wait_asynccnt 0x1" ::: "memory");  // t's load done
      } else {
        asm volatile("s_wait_asynccnt 0x0" ::: "memory");
      }
    }
    __syncthreads();
    const float* Bs = &BC[t & 1][0];
    const float* Cs = &BC[t & 1][64];
    const int row = b * SEQ_ + t;
    const float dt = delta[(size_t)row * DIN_ + d];
    const float u  = xs[(size_t)row * DIN_ + d];
    const float du = dt * u;
    float ssum = 0.f;
#pragma unroll
    for (int n = 0; n < NS_; ++n) {
      const float dA = __expf(dt * A_[n]);
      st[n] = dA * st[n] + du * Bs[n];
      ssum += st[n] * Cs[n];
    }
    y[(size_t)row * DIN_ + d] = ssum + u * Dd;
    __syncthreads();                        // protect BC[t&1] before t+2 issue
  }
}

// g = y * silu(res), res = second half of xz columns
__global__ void gate_mul(const float* __restrict__ y, const float* __restrict__ xz,
                         float* __restrict__ g) {
  const int idx = blockIdx.x * blockDim.x + threadIdx.x;
  if (idx >= ROWS_ * DIN_) return;
  const int d = idx & (DIN_ - 1);
  const int row = idx >> 10;
  const float r = xz[(size_t)row * (2 * DIN_) + DIN_ + d];
  g[idx] = y[idx] * (r / (1.f + expf(-r)));
}

// out[b,t,v] = logits[b*90 + 3t + 1, v]
__global__ void gather_out(const float* __restrict__ logits, float* __restrict__ out) {
  const int idx = blockIdx.x * blockDim.x + threadIdx.x;
  if (idx >= B_ * T_ * V_) return;
  const int v = idx % V_;
  const int r = idx / V_;
  const int t = r % T_, b = r / T_;
  out[idx] = logits[(size_t)(b * SEQ_ + 3 * t + 1) * V_ + v];
}

// ---------------------------------------------------------------------------
// Host launcher
// ---------------------------------------------------------------------------
extern "C" void kernel_launch(void* const* d_in, const int* in_sizes, int n_in,
                              void* d_out, int out_size, void* d_ws, size_t ws_size,
                              hipStream_t stream) {
  const float* states  = (const float*)d_in[0];
  const int*   actions = (const int*)  d_in[1];
  const float* rtgs    = (const float*)d_in[2];
  const float* c1w = (const float*)d_in[3];  const float* c1b = (const float*)d_in[4];
  const float* c2w = (const float*)d_in[5];  const float* c2b = (const float*)d_in[6];
  const float* c3w = (const float*)d_in[7];  const float* c3b = (const float*)d_in[8];
  const float* fcw = (const float*)d_in[9];  const float* fcb = (const float*)d_in[10];
  const float* retw = (const float*)d_in[11]; const float* retb = (const float*)d_in[12];
  const float* act_emb = (const float*)d_in[13];
  const float* embw = (const float*)d_in[14]; const float* embb = (const float*)d_in[15];
  const float* normw = (const float*)d_in[16];
  const float* ipw  = (const float*)d_in[17];
  const float* cw   = (const float*)d_in[18]; const float* cb = (const float*)d_in[19];
  const float* xpw  = (const float*)d_in[20];
  const float* dtw  = (const float*)d_in[21]; const float* dtb = (const float*)d_in[22];
  const float* alog = (const float*)d_in[23]; const float* Dp  = (const float*)d_in[24];
  const float* opw  = (const float*)d_in[25];
  const float* normfw = (const float*)d_in[26];
  const float* headw  = (const float*)d_in[27];
  float* out = (float*)d_out;

  float* ws = (float*)d_ws;
  float* state_e = ws;                         // 480*512 = 245760
  float* R = ws + 245760;
  // Phase A (encoder)
  float* h1 = R;                               // 480*32*20*20 = 6,144,000
  float* h2 = R + 6144000;                     // 480*64*9*9   = 2,488,320
  float* h3 = R + 8632320;                     // 480*64*7*7   = 1,505,280
  // Phase B (reuses R)
  float* x     = R;                            // 1440*512
  float* xn    = R + 737280;                   // 1440*512
  float* xz    = R + 1474560;                  // 1440*2048
  float* xs    = R + 4423680;                  // 1440*1024
  float* xdbl  = R + 5898240;                  // 1440*160
  float* delta = R + 6128640;                  // 1440*1024
  float* ybuf  = R + 7603200;                  // 1440*1024
  float* tok   = ybuf;                         // alias (pre-layer use only)
  float* logits = xz;                          // alias (post-layer use only)

  dim3 blk(32, 4);

  // --- Encoder: implicit-GEMM convs + FC ---
  conv_wmma<2><<<dim3(1, 3000), blk, 0, stream>>>(states, c1w, c1b, h1,
      4, 84, 84, 32, 20, 20, 8, 4, 256, BT_ * 400);
  conv_wmma<4><<<dim3(1, 608),  blk, 0, stream>>>(h1, c2w, c2b, h2,
      32, 20, 20, 64, 9, 9, 4, 2, 512, BT_ * 81);
  conv_wmma<4><<<dim3(1, 368),  blk, 0, stream>>>(h2, c3w, c3b, h3,
      64, 9, 9, 64, 7, 7, 3, 1, 576, BT_ * 49);
  gemm_wmma<ACT_TANH, true, false><<<dim3(8, 8), blk, 0, stream>>>(
      h3, 3136, fcw, 512, state_e, 512, fcb, BT_, 512, 3136);

  // --- Tokens + embedding ---
  build_tokens<<<(ROWS_ * NE_ + 255) / 256, 256, 0, stream>>>(
      rtgs, actions, state_e, retw, retb, act_emb, tok);
  gemm_wmma<ACT_NONE, true, false><<<dim3(8, 23), blk, 0, stream>>>(
      tok, 512, embw, 512, x, 512, embb, ROWS_, 512, 512);

  // --- Mamba layers ---
  for (int i = 0; i < L_; ++i) {
    rmsnorm_k<<<ROWS_, 128, 0, stream>>>(x, normw + i * DM_, xn);
    gemm_wmma<ACT_NONE, false, false><<<dim3(32, 23), blk, 0, stream>>>(
        xn, 512, ipw + (size_t)i * DM_ * 2 * DIN_, 2 * DIN_, xz, 2 * DIN_,
        nullptr, ROWS_, 2 * DIN_, DM_);
    dwconv_silu<<<(ROWS_ * DIN_ + 255) / 256, 256, 0, stream>>>(
        xz, cw + (size_t)i * DIN_ * 4, cb + (size_t)i * DIN_, xs);
    gemm_wmma<ACT_NONE, false, false><<<dim3(3, 23), blk, 0, stream>>>(
        xs, DIN_, xpw + (size_t)i * DIN_ * 160, 160, xdbl, 160,
        nullptr, ROWS_, 160, DIN_);
    gemm_wmma<ACT_SOFTPLUS, true, false><<<dim3(16, 23), blk, 0, stream>>>(
        xdbl, 160, dtw + (size_t)i * DTR_ * DIN_, DIN_, delta, DIN_,
        dtb + (size_t)i * DIN_, ROWS_, DIN_, DTR_);
    sscan<<<dim3(8, 16), 128, 0, stream>>>(
        xs, delta, xdbl, alog + (size_t)i * DIN_ * NS_, Dp + (size_t)i * DIN_, ybuf);
    gate_mul<<<(ROWS_ * DIN_ + 255) / 256, 256, 0, stream>>>(ybuf, xz, xs);
    gemm_wmma<ACT_NONE, false, true><<<dim3(8, 23), blk, 0, stream>>>(
        xs, DIN_, opw + (size_t)i * DIN_ * DM_, DM_, x, DM_,
        nullptr, ROWS_, DM_, DIN_);   // residual add into x
  }

  // --- Final norm + head + output gather ---
  rmsnorm_k<<<ROWS_, 128, 0, stream>>>(x, normfw, xn);
  gemm_wmma<ACT_NONE, false, false><<<dim3(1, 23), blk, 0, stream>>>(
      xn, 512, headw, V_, logits, V_, nullptr, ROWS_, V_, 512);
  gather_out<<<(B_ * T_ * V_ + 255) / 256, 256, 0, stream>>>(logits, out);
}
